// PointNetEncoder_52664888984072
// MI455X (gfx1250) — matrix operand order repaired and checked by reading, not compiled, not run on hardware
//
#include <hip/hip_runtime.h>
#include <hip/hip_bf16.h>
#include <math.h>

typedef __attribute__((ext_vector_type(16))) __bf16 v16bf;
typedef __attribute__((ext_vector_type(8)))  float  v8f;
typedef unsigned int   u32;
typedef unsigned short u16;

#define NBATCH 16
#define NPTS   65536
#define NPTOT  (NBATCH * NPTS)
#define NBUCK  11

// ---- workspace layout (units: 4-byte words) ----
#define OFF_SX      0      // 3 (+1 pad)
#define OFF_SXX     4      // 6 (+2 pad)
#define OFF_YENCMIN 12     // 16 (u32 encoded)
#define OFF_YENCMAX 28     // 16
#define OFF_YMIN    44     // 16
#define OFF_INVR    60     // 16
#define OFF_BSX     76     // 176
#define OFF_BSZ     252    // 176
#define OFF_BCNT    428    // 176
#define OFF_CX      604    // 176
#define OFF_CZ      780    // 176
#define OFF_SSUM2   956    // 128
#define OFF_SSQ2    1084   // 128
#define OFF_SSUM3   1212   // 256
#define OFF_SSQ3    1468   // 256
#define OFF_W1E     1724   // 192 (3x64, [i][c] = i*64+c)
#define OFF_B1E     1916   // 64
#define OFF_B2E     1980   // 128
#define OFF_B3E     2108   // 256
#define OFF_SC2     2364   // 128
#define OFF_SC3     2492   // 256
#define OFF_FSUM    2748   // 16*259 = 4144
#define OFF_FMAX    6892   // 4144 (u32 encoded)
#define OFF_W2BF    11036  // 8192 bf16 = 4096 words (raw w2)
#define OFF_W3BF    15132  // 32768 bf16 = 16384 words (raw w3)
#define OFF_W2EBF   31516  // 4096 words (BN-folded w2)
#define OFF_W3EBF   35612  // 16384 words (BN-folded w3)
#define ZERO_WORDS  11036

// ---- helpers ----
__device__ __forceinline__ u32 encf(float f) {
  u32 u = __float_as_uint(f);
  return u ^ ((u32)((int)u >> 31) | 0x80000000u);
}
__device__ __forceinline__ float decf(u32 e) {
  u32 u = (e & 0x80000000u) ? (e ^ 0x80000000u) : ~e;
  return __uint_as_float(u);
}
__device__ __forceinline__ u16 f2bf(float f) {
  u32 u = __float_as_uint(f);
  return (u16)((u + 0x7FFFu + ((u >> 16) & 1u)) >> 16);
}

union ABu { v16bf v; u32 u[8]; };

// A fragment (16x32 bf16 tile) from LDS row of one point.
// lane: row M = lane&15, khalf = lane>>4.  Element e -> K = (e<8?e:e+8)+8*khalf (+32*kc).
__device__ __forceinline__ v16bf load_a_lds(const u16* row, int kc, int khalf) {
  ABu r;
  const u32* p = (const u32*)(row + kc * 32 + khalf * 8);
  r.u[0] = p[0];  r.u[1] = p[1];  r.u[2] = p[2];  r.u[3] = p[3];
  r.u[4] = p[8];  r.u[5] = p[9];  r.u[6] = p[10]; r.u[7] = p[11];
  return r.v;
}
// B fragment (32x16 bf16 tile): lane column N = n0 + (lane&15), khalf = lane>>4.
// Element e -> K = (e<8?e:e+8)+8*khalf (+kbase), matching the A fragment.
__device__ __forceinline__ v16bf load_b_gl(const u16* w, int ncols, int kbase, int n0, int lane) {
  ABu r;
  const u16* p = w + (size_t)kbase * ncols + n0 + (lane & 15);
  int kh8 = (lane >> 4) * 8;
#pragma unroll
  for (int j = 0; j < 8; ++j) {
    int k = kh8 + 2 * j + (j >= 4 ? 8 : 0);
    r.u[j] = (u32)p[(size_t)k * ncols] | ((u32)p[(size_t)(k + 1) * ncols] << 16);
  }
  return r.v;
}
__device__ __forceinline__ v8f wmma_bf16(v16bf a, v16bf b, v8f c) {
  return __builtin_amdgcn_wmma_f32_16x16x32_bf16(false, a, false, b, (short)0, c, false, false);
}

// h1 = relu(x @ w1e + b1e), 3->64, two lanes per point (hf = half), packed bf16 to LDS.
__device__ __forceinline__ void h1_to_lds(const float* xp, const float* w1e, const float* b1e,
                                          u16* dst, int hf) {
  float x0 = xp[0], x1 = xp[1], x2 = xp[2];
#pragma unroll
  for (int j = 0; j < 16; ++j) {
    int c = hf * 32 + 2 * j;
    float a0 = fmaf(x0, w1e[c],     fmaf(x1, w1e[64 + c],     fmaf(x2, w1e[128 + c],     b1e[c])));
    float a1 = fmaf(x0, w1e[c + 1], fmaf(x1, w1e[64 + c + 1], fmaf(x2, w1e[128 + c + 1], b1e[c + 1])));
    a0 = a0 > 0.f ? a0 : 0.f;
    a1 = a1 > 0.f ? a1 : 0.f;
    ((u32*)dst)[hf * 16 + j] = (u32)f2bf(a0) | ((u32)f2bf(a1) << 16);
  }
}

// ================= kernels =================

__global__ __launch_bounds__(256) void k0_init(u32* w) {
  for (int i = blockIdx.x * 256 + threadIdx.x; i < ZERO_WORDS; i += gridDim.x * 256)
    w[i] = (i >= OFF_YENCMIN && i < OFF_YENCMIN + 16) ? 0xFFFFFFFFu : 0u;
}

__global__ __launch_bounds__(256) void k1_moments(const float* __restrict__ x, u32* ws_u, float* ws_f) {
  __shared__ float sred[9];
  __shared__ u32 symn, symx;
  int tid = threadIdx.x, b = blockIdx.y;
  if (tid < 9) sred[tid] = 0.f;
  if (tid == 0) { symn = 0xFFFFFFFFu; symx = 0u; }
  __syncthreads();
  float a[9] = {0, 0, 0, 0, 0, 0, 0, 0, 0};
  float mn = 1e30f, mx = -1e30f;
  size_t base = (size_t)b * NPTS + (size_t)blockIdx.x * 2048;
#pragma unroll
  for (int it = 0; it < 8; ++it) {
    const float* xp = x + (base + it * 256 + tid) * 3;
    float x0 = xp[0], x1 = xp[1], x2 = xp[2];
    a[0] += x0; a[1] += x1; a[2] += x2;
    a[3] += x0 * x0; a[4] += x0 * x1; a[5] += x0 * x2;
    a[6] += x1 * x1; a[7] += x1 * x2; a[8] += x2 * x2;
    mn = fminf(mn, x1); mx = fmaxf(mx, x1);
  }
#pragma unroll
  for (int j = 0; j < 9; ++j) atomicAdd(&sred[j], a[j]);
  atomicMin(&symn, encf(mn));
  atomicMax(&symx, encf(mx));
  __syncthreads();
  if (tid < 9) {
    float* dst = (tid < 3) ? (ws_f + OFF_SX + tid) : (ws_f + OFF_SXX + (tid - 3));
    atomicAdd(dst, sred[tid]);
  }
  if (tid == 0) {
    atomicMin(&ws_u[OFF_YENCMIN + b], symn);
    atomicMax(&ws_u[OFF_YENCMAX + b], symx);
  }
}

__global__ __launch_bounds__(256) void k2_fin1(const float* w1, const float* b1, const float* g1,
                                               const float* be1, const float* w2, const float* w3,
                                               float* ws_f, u32* ws_u) {
  int tid = threadIdx.x;
  const float inv = 1.0f / (float)NPTOT;
  if (tid < 64) {
    float e0 = ws_f[OFF_SX + 0] * inv, e1 = ws_f[OFF_SX + 1] * inv, e2 = ws_f[OFF_SX + 2] * inv;
    float s00 = ws_f[OFF_SXX + 0] * inv, s01 = ws_f[OFF_SXX + 1] * inv, s02 = ws_f[OFF_SXX + 2] * inv;
    float s11 = ws_f[OFF_SXX + 3] * inv, s12 = ws_f[OFF_SXX + 4] * inv, s22 = ws_f[OFF_SXX + 5] * inv;
    float wa = w1[0 * 64 + tid], wb = w1[1 * 64 + tid], wc = w1[2 * 64 + tid];
    float bb = b1[tid];
    float aa = wa * e0 + wb * e1 + wc * e2;
    float q  = wa * wa * s00 + wb * wb * s11 + wc * wc * s22
             + 2.f * (wa * wb * s01 + wa * wc * s02 + wb * wc * s12);
    float m = aa + bb;
    float v = (q + 2.f * bb * aa + bb * bb) - m * m;
    float s = rsqrtf(v + 1e-5f) * g1[tid];
    float t = be1[tid] - m * s;
    ws_f[OFF_W1E + 0 * 64 + tid] = wa * s;
    ws_f[OFF_W1E + 1 * 64 + tid] = wb * s;
    ws_f[OFF_W1E + 2 * 64 + tid] = wc * s;
    ws_f[OFF_B1E + tid] = bb * s + t;
  }
  if (tid < 16) {
    float mn = decf(ws_u[OFF_YENCMIN + tid]);
    float mx = decf(ws_u[OFF_YENCMAX + tid]);
    ws_f[OFF_YMIN + tid] = mn;
    ws_f[OFF_INVR + tid] = 1.f / (mx - mn + 1e-6f);
  }
  u16* w2bf = (u16*)(ws_u + OFF_W2BF);
  for (int i = tid; i < 64 * 128; i += 256) w2bf[i] = f2bf(w2[i]);
  u16* w3bf = (u16*)(ws_u + OFF_W3BF);
  for (int i = tid; i < 128 * 256; i += 256) w3bf[i] = f2bf(w3[i]);
}

__global__ __launch_bounds__(256) void k3_bucket(const float* __restrict__ x, const float* ws_f,
                                                 float* bsx, float* bsz, float* bcnt) {
  __shared__ float lx[NBUCK], lz[NBUCK];
  __shared__ int lc[NBUCK];
  int tid = threadIdx.x, b = blockIdx.y;
  if (tid < NBUCK) { lx[tid] = 0.f; lz[tid] = 0.f; lc[tid] = 0; }
  __syncthreads();
  float ymin = ws_f[OFF_YMIN + b], invr = ws_f[OFF_INVR + b];
  size_t base = (size_t)b * NPTS + (size_t)blockIdx.x * 2048;
#pragma unroll
  for (int it = 0; it < 8; ++it) {
    const float* xp = x + (base + it * 256 + tid) * 3;
    float x0 = xp[0], yv = xp[1], z0 = xp[2];
    int bk = (int)((yv - ymin) * invr * 10.f);
    bk = bk < 0 ? 0 : (bk > NBUCK - 1 ? NBUCK - 1 : bk);
    atomicAdd(&lx[bk], x0);
    atomicAdd(&lz[bk], z0);
    atomicAdd(&lc[bk], 1);
  }
  __syncthreads();
  if (tid < NBUCK) {
    atomicAdd(&bsx[b * NBUCK + tid], lx[tid]);
    atomicAdd(&bsz[b * NBUCK + tid], lz[tid]);
    atomicAdd(&bcnt[b * NBUCK + tid], (float)lc[tid]);
  }
}

__global__ __launch_bounds__(256) void k4_centers(float* ws_f) {
  int i = threadIdx.x;
  if (i < NBATCH * NBUCK) {
    float d = fmaxf(ws_f[OFF_BCNT + i], 1.f);
    ws_f[OFF_CX + i] = ws_f[OFF_BSX + i] / d;
    ws_f[OFF_CZ + i] = ws_f[OFF_BSZ + i] / d;
  }
}

// layer-2 pre-activation stats: a2 = h1 @ w2 + b2 (raw weights)
__global__ __launch_bounds__(256) void k5_stats2(const float* __restrict__ x, const float* ws_f,
                                                 const u16* __restrict__ w2bf, const float* b2,
                                                 float* gsum, float* gssq) {
  __shared__ u16 h1s[8][16][64];
  __shared__ float ssum[128], ssq[128];
  int tid = threadIdx.x, wv = tid >> 5, ln = tid & 31;
  if (tid < 128) { ssum[tid] = 0.f; ssq[tid] = 0.f; }
  __syncthreads();
  const float* w1e = ws_f + OFF_W1E;
  const float* b1e = ws_f + OFF_B1E;
  int row = ln & 15, kh = ln >> 4;
  // preload this lane's per-channel biases (channel = nt*16 + (ln&15))
  float bb2r[8];
#pragma unroll
  for (int nt = 0; nt < 8; ++nt) bb2r[nt] = b2[nt * 16 + (ln & 15)];
  for (int it = 0; it < 8; ++it) {
    size_t base = (size_t)blockIdx.x * 1024 + it * 128 + wv * 16;
    h1_to_lds(x + (base + (ln >> 1)) * 3, w1e, b1e, &h1s[wv][ln >> 1][0], ln & 1);
    __syncthreads();
    v16bf a0 = load_a_lds(&h1s[wv][row][0], 0, kh);
    v16bf a1 = load_a_lds(&h1s[wv][row][0], 1, kh);
#pragma unroll
    for (int nt = 0; nt < 8; ++nt) {
      v8f acc = {};
      acc = wmma_bf16(a0, load_b_gl(w2bf, 128, 0, nt * 16, ln), acc);
      acc = wmma_bf16(a1, load_b_gl(w2bf, 128, 32, nt * 16, ln), acc);
      int ch = nt * 16 + (ln & 15);
      float bb = bb2r[nt], s = 0.f, q = 0.f;
#pragma unroll
      for (int r = 0; r < 8; ++r) { float v = acc[r] + bb; s += v; q += v * v; }
      s += __shfl_xor(s, 16, 32);
      q += __shfl_xor(q, 16, 32);
      if (ln < 16) { atomicAdd(&ssum[ch], s); atomicAdd(&ssq[ch], q); }
    }
    __syncthreads();
  }
  if (tid < 128) { atomicAdd(&gsum[tid], ssum[tid]); atomicAdd(&gssq[tid], ssq[tid]); }
}

__global__ __launch_bounds__(256) void k6_fin2(const float* w2, const float* b2, const float* g2,
                                               const float* be2, float* ws_f, u32* ws_u) {
  int tid = threadIdx.x;
  const float inv = 1.0f / (float)NPTOT;
  if (tid < 128) {
    float m = ws_f[OFF_SSUM2 + tid] * inv;
    float v = ws_f[OFF_SSQ2 + tid] * inv - m * m;
    float s = rsqrtf(v + 1e-5f) * g2[tid];
    ws_f[OFF_SC2 + tid] = s;
    ws_f[OFF_B2E + tid] = b2[tid] * s + (be2[tid] - m * s);
  }
  __syncthreads();
  u16* w2e = (u16*)(ws_u + OFF_W2EBF);
  for (int i = tid; i < 64 * 128; i += 256)
    w2e[i] = f2bf(w2[i] * ws_f[OFF_SC2 + (i & 127)]);
}

// layer-3 pre-activation stats: h2 via folded w2e, a3 = h2 @ w3 + b3 (raw)
__global__ __launch_bounds__(256) void k7_stats3(const float* __restrict__ x, const float* ws_f,
                                                 const u16* __restrict__ w2e, const u16* __restrict__ w3bf,
                                                 const float* b3, float* gsum, float* gssq) {
  __shared__ u16 h1s[8][16][64];
  __shared__ u16 h2s[8][16][128];
  __shared__ float ssum[256], ssq[256];
  int tid = threadIdx.x, wv = tid >> 5, ln = tid & 31;
  ssum[tid] = 0.f; ssq[tid] = 0.f;
  __syncthreads();
  const float* w1e = ws_f + OFF_W1E;
  const float* b1e = ws_f + OFF_B1E;
  const float* b2e = ws_f + OFF_B2E;
  int row = ln & 15, kh = ln >> 4;
  float bb2r[8], bb3r[16];
#pragma unroll
  for (int nt = 0; nt < 8; ++nt) bb2r[nt] = b2e[nt * 16 + (ln & 15)];
#pragma unroll
  for (int nt = 0; nt < 16; ++nt) bb3r[nt] = b3[nt * 16 + (ln & 15)];
  for (int it = 0; it < 8; ++it) {
    size_t base = (size_t)blockIdx.x * 1024 + it * 128 + wv * 16;
    h1_to_lds(x + (base + (ln >> 1)) * 3, w1e, b1e, &h1s[wv][ln >> 1][0], ln & 1);
    __syncthreads();
    v16bf a0 = load_a_lds(&h1s[wv][row][0], 0, kh);
    v16bf a1 = load_a_lds(&h1s[wv][row][0], 1, kh);
#pragma unroll
    for (int nt = 0; nt < 8; ++nt) {
      v8f acc = {};
      acc = wmma_bf16(a0, load_b_gl(w2e, 128, 0, nt * 16, ln), acc);
      acc = wmma_bf16(a1, load_b_gl(w2e, 128, 32, nt * 16, ln), acc);
      int ch = nt * 16 + (ln & 15);
      float bb = bb2r[nt];
#pragma unroll
      for (int r = 0; r < 8; ++r) {
        float v = acc[r] + bb;
        v = v > 0.f ? v : 0.f;
        h2s[wv][r + 8 * kh][ch] = f2bf(v);
      }
    }
    __syncthreads();
    v16bf c0 = load_a_lds(&h2s[wv][row][0], 0, kh);
    v16bf c1 = load_a_lds(&h2s[wv][row][0], 1, kh);
    v16bf c2 = load_a_lds(&h2s[wv][row][0], 2, kh);
    v16bf c3 = load_a_lds(&h2s[wv][row][0], 3, kh);
#pragma unroll
    for (int nt = 0; nt < 16; ++nt) {
      v8f acc = {};
      acc = wmma_bf16(c0, load_b_gl(w3bf, 256, 0,  nt * 16, ln), acc);
      acc = wmma_bf16(c1, load_b_gl(w3bf, 256, 32, nt * 16, ln), acc);
      acc = wmma_bf16(c2, load_b_gl(w3bf, 256, 64, nt * 16, ln), acc);
      acc = wmma_bf16(c3, load_b_gl(w3bf, 256, 96, nt * 16, ln), acc);
      int ch = nt * 16 + (ln & 15);
      float bb = bb3r[nt], s = 0.f, q = 0.f;
#pragma unroll
      for (int r = 0; r < 8; ++r) { float v = acc[r] + bb; s += v; q += v * v; }
      s += __shfl_xor(s, 16, 32);
      q += __shfl_xor(q, 16, 32);
      if (ln < 16) { atomicAdd(&ssum[ch], s); atomicAdd(&ssq[ch], q); }
    }
    __syncthreads();
  }
  atomicAdd(&gsum[tid], ssum[tid]);
  atomicAdd(&gssq[tid], ssq[tid]);
}

__global__ __launch_bounds__(256) void k8_fin3(const float* w3, const float* b3, const float* g3,
                                               const float* be3, float* ws_f, u32* ws_u) {
  int tid = threadIdx.x;
  const float inv = 1.0f / (float)NPTOT;
  if (tid < 256) {
    float m = ws_f[OFF_SSUM3 + tid] * inv;
    float v = ws_f[OFF_SSQ3 + tid] * inv - m * m;
    float s = rsqrtf(v + 1e-5f) * g3[tid];
    ws_f[OFF_SC3 + tid] = s;
    ws_f[OFF_B3E + tid] = b3[tid] * s + (be3[tid] - m * s);
  }
  __syncthreads();
  u16* w3e = (u16*)(ws_u + OFF_W3EBF);
  for (int i = tid; i < 128 * 256; i += 256)
    w3e[i] = f2bf(w3[i] * ws_f[OFF_SC3 + (i & 255)]);
}

// fused final pass: h1 -> h2 -> feat256 (+spatial), per-batch max/mean reduction
__global__ __launch_bounds__(256) void k9_feat(const float* __restrict__ x, const float* ws_f,
                                               const u16* __restrict__ w2e, const u16* __restrict__ w3e,
                                               float* fsum, u32* fmaxe) {
  __shared__ u16 h1s[8][16][64];
  __shared__ u16 h2s[8][16][128];
  __shared__ float ssum[259];
  __shared__ u32 smax[259];
  int tid = threadIdx.x, wv = tid >> 5, ln = tid & 31;
  int b = blockIdx.y;
  for (int i = tid; i < 259; i += 256) { ssum[i] = 0.f; smax[i] = 0u; }
  __syncthreads();
  const float* w1e = ws_f + OFF_W1E;
  const float* b1e = ws_f + OFF_B1E;
  const float* b2e = ws_f + OFF_B2E;
  const float* b3e = ws_f + OFF_B3E;
  const float* cx = ws_f + OFF_CX + b * NBUCK;
  const float* cz = ws_f + OFF_CZ + b * NBUCK;
  float ymin = ws_f[OFF_YMIN + b], invr = ws_f[OFF_INVR + b];
  int row = ln & 15, kh = ln >> 4;
  float bb2r[8], bb3r[16];
#pragma unroll
  for (int nt = 0; nt < 8; ++nt) bb2r[nt] = b2e[nt * 16 + (ln & 15)];
#pragma unroll
  for (int nt = 0; nt < 16; ++nt) bb3r[nt] = b3e[nt * 16 + (ln & 15)];
  for (int it = 0; it < 8; ++it) {
    size_t n = (size_t)blockIdx.x * 1024 + it * 128 + wv * 16;
    size_t pt = (size_t)b * NPTS + n;
    h1_to_lds(x + (pt + (ln >> 1)) * 3, w1e, b1e, &h1s[wv][ln >> 1][0], ln & 1);
    __syncthreads();
    v16bf a0 = load_a_lds(&h1s[wv][row][0], 0, kh);
    v16bf a1 = load_a_lds(&h1s[wv][row][0], 1, kh);
#pragma unroll
    for (int nt = 0; nt < 8; ++nt) {
      v8f acc = {};
      acc = wmma_bf16(a0, load_b_gl(w2e, 128, 0, nt * 16, ln), acc);
      acc = wmma_bf16(a1, load_b_gl(w2e, 128, 32, nt * 16, ln), acc);
      int ch = nt * 16 + (ln & 15);
      float bb = bb2r[nt];
#pragma unroll
      for (int r = 0; r < 8; ++r) {
        float v = acc[r] + bb;
        v = v > 0.f ? v : 0.f;
        h2s[wv][r + 8 * kh][ch] = f2bf(v);
      }
    }
    __syncthreads();
    v16bf c0 = load_a_lds(&h2s[wv][row][0], 0, kh);
    v16bf c1 = load_a_lds(&h2s[wv][row][0], 1, kh);
    v16bf c2 = load_a_lds(&h2s[wv][row][0], 2, kh);
    v16bf c3 = load_a_lds(&h2s[wv][row][0], 3, kh);
#pragma unroll
    for (int nt = 0; nt < 16; ++nt) {
      v8f acc = {};
      acc = wmma_bf16(c0, load_b_gl(w3e, 256, 0,  nt * 16, ln), acc);
      acc = wmma_bf16(c1, load_b_gl(w3e, 256, 32, nt * 16, ln), acc);
      acc = wmma_bf16(c2, load_b_gl(w3e, 256, 64, nt * 16, ln), acc);
      acc = wmma_bf16(c3, load_b_gl(w3e, 256, 96, nt * 16, ln), acc);
      int ch = nt * 16 + (ln & 15);
      float bb = bb3r[nt];
      float lmax = -1e30f, ls = 0.f;
#pragma unroll
      for (int r = 0; r < 8; ++r) {
        float v = acc[r] + bb;
        v = v > 0.f ? v : 0.f;
        lmax = fmaxf(lmax, v);
        ls += v;
      }
      lmax = fmaxf(lmax, __shfl_xor(lmax, 16, 32));
      ls += __shfl_xor(ls, 16, 32);
      if (ln < 16) { atomicMax(&smax[ch], encf(lmax)); atomicAdd(&ssum[ch], ls); }
    }
    // spatial features (3 channels) — lanes 0..15 handle the wave's 16 points
    if (ln < 16) {
      const float* xp = x + (pt + ln) * 3;
      float x0 = xp[0], yv = xp[1], z0 = xp[2];
      int bk = (int)((yv - ymin) * invr * 10.f);
      bk = bk < 0 ? 0 : (bk > NBUCK - 1 ? NBUCK - 1 : bk);
      float dx = x0 - cx[bk], dz = z0 - cz[bk];
      float r = sqrtf(dx * dx + dz * dz);
      float inv = r > 0.f ? 1.f / r : 0.f;
      float sn = dz * inv;
      float cs = r > 0.f ? dx * inv : 1.f;
      atomicMax(&smax[256], encf(sn)); atomicAdd(&ssum[256], sn);
      atomicMax(&smax[257], encf(cs)); atomicAdd(&ssum[257], cs);
      atomicMax(&smax[258], encf(r));  atomicAdd(&ssum[258], r);
    }
    __syncthreads();
  }
  for (int i = tid; i < 259; i += 256) {
    atomicMax(&fmaxe[b * 259 + i], smax[i]);
    atomicAdd(&fsum[b * 259 + i], ssum[i]);
  }
}

// head: featcat(518) @ wp(518x512) + bp, LayerNorm over 512
__global__ __launch_bounds__(512) void k10_head(const u32* fmaxe, const float* fsum,
                                                const float* __restrict__ wp, const float* bp,
                                                const float* gl, const float* bl, float* out) {
  __shared__ float fc[518];
  __shared__ float s1[512], s2[512];
  int tid = threadIdx.x;
  for (int b = 0; b < NBATCH; ++b) {
    if (tid < 259) {
      fc[tid] = decf(fmaxe[b * 259 + tid]);
      fc[259 + tid] = fsum[b * 259 + tid] * (1.f / (float)NPTS);
    }
    __syncthreads();
    float o = bp[tid];
    for (int j = 0; j < 518; ++j) o = fmaf(fc[j], wp[j * 512 + tid], o);
    s1[tid] = o; s2[tid] = o * o;
    __syncthreads();
    for (int off = 256; off > 0; off >>= 1) {
      if (tid < off) { s1[tid] += s1[tid + off]; s2[tid] += s2[tid + off]; }
      __syncthreads();
    }
    float mu = s1[0] * (1.f / 512.f);
    float var = s2[0] * (1.f / 512.f) - mu * mu;
    out[b * 512 + tid] = (o - mu) * rsqrtf(var + 1e-5f) * gl[tid] + bl[tid];
    __syncthreads();
  }
}

extern "C" void kernel_launch(void* const* d_in, const int* in_sizes, int n_in,
                              void* d_out, int out_size, void* d_ws, size_t ws_size,
                              hipStream_t stream) {
  const float* x   = (const float*)d_in[0];
  const float* w1  = (const float*)d_in[1];
  const float* b1  = (const float*)d_in[2];
  const float* g1  = (const float*)d_in[3];
  const float* be1 = (const float*)d_in[4];
  const float* w2  = (const float*)d_in[5];
  const float* b2  = (const float*)d_in[6];
  const float* g2  = (const float*)d_in[7];
  const float* be2 = (const float*)d_in[8];
  const float* w3  = (const float*)d_in[9];
  const float* b3  = (const float*)d_in[10];
  const float* g3  = (const float*)d_in[11];
  const float* be3 = (const float*)d_in[12];
  const float* wp  = (const float*)d_in[13];
  const float* bp  = (const float*)d_in[14];
  const float* gln = (const float*)d_in[15];
  const float* bln = (const float*)d_in[16];

  float* ws_f = (float*)d_ws;
  u32*   ws_u = (u32*)d_ws;
  const u16* w2bf  = (const u16*)(ws_u + OFF_W2BF);
  const u16* w3bf  = (const u16*)(ws_u + OFF_W3BF);
  const u16* w2ebf = (const u16*)(ws_u + OFF_W2EBF);
  const u16* w3ebf = (const u16*)(ws_u + OFF_W3EBF);

  k0_init<<<dim3(16), 256, 0, stream>>>(ws_u);
  k1_moments<<<dim3(32, 16), 256, 0, stream>>>(x, ws_u, ws_f);
  k2_fin1<<<1, 256, 0, stream>>>(w1, b1, g1, be1, w2, w3, ws_f, ws_u);
  k3_bucket<<<dim3(32, 16), 256, 0, stream>>>(x, ws_f, ws_f + OFF_BSX, ws_f + OFF_BSZ, ws_f + OFF_BCNT);
  k4_centers<<<1, 256, 0, stream>>>(ws_f);
  k5_stats2<<<dim3(1024), 256, 0, stream>>>(x, ws_f, w2bf, b2, ws_f + OFF_SSUM2, ws_f + OFF_SSQ2);
  k6_fin2<<<1, 256, 0, stream>>>(w2, b2, g2, be2, ws_f, ws_u);
  k7_stats3<<<dim3(1024), 256, 0, stream>>>(x, ws_f, w2ebf, w3bf, b3, ws_f + OFF_SSUM3, ws_f + OFF_SSQ3);
  k8_fin3<<<1, 256, 0, stream>>>(w3, b3, g3, be3, ws_f, ws_u);
  k9_feat<<<dim3(64, 16), 256, 0, stream>>>(x, ws_f, w2ebf, w3ebf, ws_f + OFF_FSUM, ws_u + OFF_FMAX);
  k10_head<<<1, 512, 0, stream>>>(ws_u + OFF_FMAX, ws_f + OFF_FSUM, wp, bp, gln, bln, (float*)d_out);
}
